// SwinBlock_official_80831284511241
// MI455X (gfx1250) — matrix operand order repaired and checked by reading, not compiled, not run on hardware
//
#include <hip/hip_runtime.h>

// ---------------------------------------------------------------------------
// Types for CDNA5 WMMA (wave32, v_wmma_f32_16x16x32_bf16)
// ---------------------------------------------------------------------------
typedef __attribute__((ext_vector_type(16))) __bf16 bf16x16;
typedef __attribute__((ext_vector_type(8)))  float  floatx8;
typedef __attribute__((ext_vector_type(4)))  unsigned u32x4;

constexpr int HH = 128, WW = 128, CC = 256;
constexpr int SHIFT = 4;
constexpr int NWINDOWS = 2048;            // B * 16 * 16
constexpr int NTOKENS  = NWINDOWS * 64;   // 131072

#ifndef __has_builtin
#define __has_builtin(x) 0
#endif

// ---------------------------------------------------------------------------
// Tensor Data Mover (TDM) staging, guarded so the file compiles on toolchains
// without the builtin (falls back to vectorized global->LDS copies).
// ---------------------------------------------------------------------------
#if __has_builtin(__builtin_amdgcn_tensor_load_to_lds)
#define HAVE_TDM 1
#else
#define HAVE_TDM 0
#endif

#if __has_builtin(__builtin_amdgcn_s_wait_tensorcnt)
#define WAIT_TENSORCNT0() __builtin_amdgcn_s_wait_tensorcnt(0)
#else
#define WAIT_TENSORCNT0() asm volatile("s_wait_tensorcnt 0x0" ::: "memory")
#endif

#if HAVE_TDM
typedef __attribute__((ext_vector_type(4))) unsigned tdm_u4;
typedef __attribute__((ext_vector_type(8))) int      tdm_i8;
typedef __attribute__((ext_vector_type(4))) int      tdm_i4;

// DMA a 2-D bf16 tile (tile_d0=32 elems per row, tile_d1 rows) from a
// row-major matrix (row stride = strideElems) into LDS, padding each 64B row
// with 16B -> effective LDS row stride of 40 bf16 (matches the frag loaders).
// D# bitfields per CDNA5 ISA 8.3/8.4 (group0: count/lds/global/type,
// group1: data_size=2B, pad_enable, pad_interval=16 DW, pad_amount=4 DW,
// tensor dims / tile dims / dim0 stride).
__device__ __forceinline__ void tdm_load_tile_bf16(const unsigned short* gsrc,
                                                   unsigned ldsByteOff,
                                                   int strideElems, int rows) {
  const unsigned long long ga = (unsigned long long)(size_t)gsrc;
  tdm_u4 g0;
  g0[0] = 1u;                                            // count=1, user mode
  g0[1] = ldsByteOff;                                    // lds_addr
  g0[2] = (unsigned)(ga & 0xFFFFFFFFu);                  // global_addr lo
  g0[3] = (unsigned)((ga >> 32) & 0x01FFFFFFu) | (2u << 30);  // addr hi | type=2
  tdm_i8 g1;
  g1[0] = (int)((1u << 16) |        // data_size = 2 bytes
                (1u << 20) |        // pad_enable
                (3u << 22) |        // pad_interval: 16 DWORDs (one 64B row)
                (3u << 25));        // pad_amount:   4 DWORDs (16B)
  g1[1] = (int)(((unsigned)strideElems & 0xFFFFu) << 16);          // tensor_dim0 lo
  g1[2] = (int)((((unsigned)strideElems >> 16) & 0xFFFFu) |
                (((unsigned)rows & 0xFFFFu) << 16));               // d0 hi | tensor_dim1 lo
  g1[3] = (int)((((unsigned)rows >> 16) & 0xFFFFu) | (32u << 16)); // d1 hi | tile_dim0=32
  g1[4] = (int)((unsigned)rows & 0xFFFFu);                         // tile_dim1, tile_dim2=0
  g1[5] = (int)(unsigned)strideElems;                              // tensor_dim0_stride lo
  g1[6] = 0;                                                       // stride hi | d1 stride
  g1[7] = 0;
  const tdm_i4 z4 = {0, 0, 0, 0};
#if defined(__clang_major__) && (__clang_major__ >= 23)
  const tdm_i8 z8 = {0, 0, 0, 0, 0, 0, 0, 0};
  __builtin_amdgcn_tensor_load_to_lds(g0, g1, z4, z4, z8, 0);
#else
  __builtin_amdgcn_tensor_load_to_lds(g0, g1, z4, z4, 0);
#endif
}

// Low 32 bits of a generic pointer to __shared__ == LDS byte address (ISA 10.2).
__device__ __forceinline__ unsigned lds_addr32(const void* p) {
  return (unsigned)(size_t)p;
}
#endif  // HAVE_TDM

__device__ __forceinline__ unsigned short f2u16(float f) {
  unsigned u = __builtin_bit_cast(unsigned, f);
  unsigned r = u + 0x7FFFu + ((u >> 16) & 1u);   // round-to-nearest-even bf16
  return (unsigned short)(r >> 16);
}

union FragU { bf16x16 v; u32x4 q[2]; };

// A-matrix 16x32 bf16 fragment (ISA 7.12.2):
// lane<16: row = row0+lane, K = {0..7, 16..23}; lane>=16: row0+lane-16, K={8..15, 24..31}
__device__ __forceinline__ bf16x16 load_fragA(const unsigned short* base, int row0,
                                              int k0, int stride, int lane) {
  const int r = row0 + (lane & 15);
  const unsigned short* p = base + r * stride + k0 + (lane >> 4) * 8;
  FragU f;
  f.q[0] = *(const u32x4*)p;          // K +0..7   (pairs)
  f.q[1] = *(const u32x4*)(p + 16);   // K +16..23
  return f.v;
}

// B-matrix 32x16 bf16 fragment, stored column-major as rows [n][k]:
// lane<16: col n = row0+lane, K = {0..7, 16..23}; lane>=16: K = {8..15, 24..31}
__device__ __forceinline__ bf16x16 load_fragB(const unsigned short* base, int row0,
                                              int k0, int stride, int lane) {
  const int n = row0 + (lane & 15);
  const unsigned short* p = base + n * stride + k0 + (lane >> 4) * 8;
  FragU f;
  f.q[0] = *(const u32x4*)p;
  f.q[1] = *(const u32x4*)(p + 16);
  return f.v;
}

__device__ __forceinline__ floatx8 wmma_bf16(bf16x16 a, bf16x16 b, floatx8 c) {
  return __builtin_amdgcn_wmma_f32_16x16x32_bf16(false, a, false, b, (short)0, c,
                                                 false, false);
}

__device__ __forceinline__ int rgn(int p) { return (p < 120) ? 0 : ((p < 124) ? 1 : 2); }

__device__ __forceinline__ float gelu_tanh(float x) {
  const float x3 = x * x * x;
  const float t = tanhf(0.7978845608028654f * (x + 0.044715f * x3));
  return 0.5f * x * (1.0f + t);
}

// ---------------------------------------------------------------------------
// fp32 -> bf16 weight conversion
// ---------------------------------------------------------------------------
__global__ __launch_bounds__(256) void cvt_bf16_kernel(const float* __restrict__ s,
                                                       unsigned short* __restrict__ d,
                                                       int n) {
  const int i = blockIdx.x * 256 + threadIdx.x;
  if (i < n) d[i] = f2u16(s[i]);
}

// ---------------------------------------------------------------------------
// LN1 + cyclic shift + window partition. One wave per token, lane over channels.
// x: (B,C,H,W) f32  ->  win: (2048 windows, 64 tok, 256 c) bf16
// ---------------------------------------------------------------------------
__global__ __launch_bounds__(256) void ln1_window_kernel(
    const float* __restrict__ x, const float* __restrict__ g,
    const float* __restrict__ b, unsigned short* __restrict__ win) {
  const int wave = threadIdx.x >> 5;
  const int lane = threadIdx.x & 31;
  const int gt = blockIdx.x * 8 + wave;       // (window, token)
  const int winId = gt >> 6, tok = gt & 63;
  const int bI = winId >> 8, wloc = winId & 255;
  const int wh = wloc >> 4, ww = wloc & 15;
  const int hs = ((wh << 3) + (tok >> 3) + SHIFT) & (HH - 1);  // roll(-SHIFT)
  const int ws = ((ww << 3) + (tok & 7) + SHIFT) & (WW - 1);
  const float* xp = x + (size_t)bI * CC * (HH * WW) + hs * WW + ws;
  float v[8];
  float s = 0.0f;
  #pragma unroll
  for (int u = 0; u < 8; ++u) {
    v[u] = xp[(size_t)(lane + 32 * u) * (HH * WW)];
    s += v[u];
  }
  #pragma unroll
  for (int o = 16; o; o >>= 1) s += __shfl_xor(s, o, 32);
  const float mu = s * (1.0f / 256.0f);
  float q = 0.0f;
  #pragma unroll
  for (int u = 0; u < 8; ++u) { const float d0 = v[u] - mu; q += d0 * d0; }
  #pragma unroll
  for (int o = 16; o; o >>= 1) q += __shfl_xor(q, o, 32);
  const float rstd = rsqrtf(q * (1.0f / 256.0f) + 1e-5f);
  unsigned short* wp = win + (size_t)gt * 256;
  #pragma unroll
  for (int u = 0; u < 8; ++u) {
    const int c = lane + 32 * u;
    wp[c] = f2u16((v[u] - mu) * rstd * g[c] + b[c]);
  }
}

// ---------------------------------------------------------------------------
// LN2: t (131072 x 256) f32 -> xn bf16
// ---------------------------------------------------------------------------
__global__ __launch_bounds__(256) void ln2_kernel(
    const float* __restrict__ t, const float* __restrict__ g,
    const float* __restrict__ b, unsigned short* __restrict__ xn) {
  const int wave = threadIdx.x >> 5;
  const int lane = threadIdx.x & 31;
  const int gt = blockIdx.x * 8 + wave;
  const float* tp = t + (size_t)gt * 256;
  float v[8];
  float s = 0.0f;
  #pragma unroll
  for (int u = 0; u < 8; ++u) { v[u] = tp[lane + 32 * u]; s += v[u]; }
  #pragma unroll
  for (int o = 16; o; o >>= 1) s += __shfl_xor(s, o, 32);
  const float mu = s * (1.0f / 256.0f);
  float q = 0.0f;
  #pragma unroll
  for (int u = 0; u < 8; ++u) { const float d0 = v[u] - mu; q += d0 * d0; }
  #pragma unroll
  for (int o = 16; o; o >>= 1) q += __shfl_xor(q, o, 32);
  const float rstd = rsqrtf(q * (1.0f / 256.0f) + 1e-5f);
  unsigned short* xp = xn + (size_t)gt * 256;
  #pragma unroll
  for (int u = 0; u < 8; ++u) {
    const int c = lane + 32 * u;
    xp[c] = f2u16((v[u] - mu) * rstd * g[c] + b[c]);
  }
}

// ---------------------------------------------------------------------------
// Generic bf16 WMMA GEMM:  out[M][N] = A[M][K] * W[N][K]^T + bias[N], fused epilogues.
// Block 256 thr = 8 waves; workgroup tile 128x128; wave tile 64x32; K chunk 32,
// double-buffered LDS. Tiles staged by the Tensor Data Mover when available
// (descriptor padding produces the stride-40 bf16 LDS rows the frag loaders use),
// otherwise by vectorized b128 copies.
// MODE 0: QKV   (K=256, N=768)  -> scale q by hd^-0.5, split to q/k/v bf16 buffers
// MODE 1: proj  (K=256, N=256)  -> window-reverse + unshift + residual(x) -> t f32
// MODE 2: fc1   (K=256, N=1024) -> gelu -> h2 bf16
// MODE 3: fc2   (K=1024,N=256)  -> + t residual -> out (B,C,H,W) f32
// ---------------------------------------------------------------------------
template <int MODE>
__global__ __launch_bounds__(256) void gemm_wmma_kernel(
    const unsigned short* __restrict__ A, const unsigned short* __restrict__ Bw,
    const float* __restrict__ bias, float* __restrict__ outF,
    unsigned short* __restrict__ outB, const float* __restrict__ res) {
  constexpr int K = (MODE == 3) ? 1024 : 256;
  constexpr int KC = K / 32;
  constexpr int STR = 40;
  __shared__ alignas(16) unsigned short As[2][128 * STR];
  __shared__ alignas(16) unsigned short Bs[2][128 * STR];

  const int tid = threadIdx.x;
  const int lane = tid & 31;
  const int wv = tid >> 5;
  const int wm = wv & 1;                    // M half
  const int wn = wv >> 1;                   // N quarter
  const int rowBase = blockIdx.y * 128;
  const int colBase = blockIdx.x * 128;

  floatx8 acc[4][2] = {};

  auto stage = [&](int buf, int kc) {
    const int k0 = kc * 32;
#if HAVE_TDM
    if (wv == 0) {   // one TDM descriptor pair per block; TENSORcnt is per-wave
      tdm_load_tile_bf16(A + (size_t)rowBase * K + k0, lds_addr32(&As[buf][0]), K, 128);
      tdm_load_tile_bf16(Bw + (size_t)colBase * K + k0, lds_addr32(&Bs[buf][0]), K, 128);
    }
#else
    for (int p = tid; p < 512; p += 256) {      // 512 x 16B transfers each
      const int row = p >> 2;
      const int kq = (p & 3) * 8;
      *(u32x4*)&As[buf][row * STR + kq] =
          *(const u32x4*)(A + (size_t)(rowBase + row) * K + k0 + kq);
    }
    for (int p = tid; p < 512; p += 256) {
      const int row = p >> 2;
      const int kq = (p & 3) * 8;
      *(u32x4*)&Bs[buf][row * STR + kq] =
          *(const u32x4*)(Bw + (size_t)(colBase + row) * K + k0 + kq);
    }
#endif
  };

  stage(0, 0);
#if HAVE_TDM
  if (wv == 0) WAIT_TENSORCNT0();
#endif
  __syncthreads();
  for (int kc = 0; kc < KC; ++kc) {
    const int buf = kc & 1;
    if (kc + 1 < KC) stage(buf ^ 1, kc + 1);   // DMA next tile during compute
    bf16x16 af[4], bf[2];
    #pragma unroll
    for (int mi = 0; mi < 4; ++mi)
      af[mi] = load_fragA(&As[buf][0], wm * 64 + mi * 16, 0, STR, lane);
    #pragma unroll
    for (int ni = 0; ni < 2; ++ni)
      bf[ni] = load_fragB(&Bs[buf][0], wn * 32 + ni * 16, 0, STR, lane);
    #pragma unroll
    for (int mi = 0; mi < 4; ++mi)
      #pragma unroll
      for (int ni = 0; ni < 2; ++ni)
        acc[mi][ni] = wmma_bf16(af[mi], bf[ni], acc[mi][ni]);
#if HAVE_TDM
    if (kc + 1 < KC && wv == 0) WAIT_TENSORCNT0();
#endif
    __syncthreads();
  }

  const int hl = lane >> 4;
  const int nl = lane & 15;
  #pragma unroll
  for (int mi = 0; mi < 4; ++mi) {
    #pragma unroll
    for (int ni = 0; ni < 2; ++ni) {
      #pragma unroll
      for (int r = 0; r < 8; ++r) {
        const int M = rowBase + wm * 64 + mi * 16 + r + hl * 8;
        const int N = colBase + wn * 32 + ni * 16 + nl;
        float v = acc[mi][ni][r] + bias[N];
        if constexpr (MODE == 0) {
          if (N < 256) v *= 0.17677669529663687f;   // q * hd^-0.5
          const int sel = N >> 8, nc = N & 255;
          const int hd = nc >> 5, d = nc & 31;
          const int winId = M >> 6, tok = M & 63;
          outB[(size_t)sel * ((size_t)NTOKENS * 256) +
               (((size_t)winId * 8 + hd) * 64 + tok) * 32 + d] = f2u16(v);
        } else if constexpr (MODE == 1) {
          const int winId = M >> 6, tok = M & 63;
          const int bI = winId >> 8, wloc = winId & 255;
          const int hh = (((wloc >> 4) << 3) + (tok >> 3) + SHIFT) & (HH - 1);
          const int wg = (((wloc & 15) << 3) + (tok & 7) + SHIFT) & (WW - 1);
          const size_t pix = (size_t)hh * WW + wg;
          const float sc = res[((size_t)bI * CC + N) * (HH * WW) + pix];
          outF[((size_t)bI * (HH * WW) + pix) * CC + N] = v + sc;
        } else if constexpr (MODE == 2) {
          outB[(size_t)M * 1024 + N] = f2u16(gelu_tanh(v));
        } else {
          const int bI = M >> 14, pix = M & 16383;
          const float sc = res[(size_t)M * 256 + N];
          outF[((size_t)bI * CC + N) * (HH * WW) + pix] = v + sc;
        }
      }
    }
  }
}

// ---------------------------------------------------------------------------
// Windowed attention: one 4-wave block per (window, head).
// scores = q@k^T (K=32 -> 1 WMMA per 16x16 tile), + rel-pos bias + shift mask,
// softmax via __shfl_xor within 16-lane halves, probs -> LDS, P@V via WMMA.
// q/k tiles staged via TDM when available.
// ---------------------------------------------------------------------------
__global__ __launch_bounds__(128) void attn_kernel(
    const unsigned short* __restrict__ qbuf, const unsigned short* __restrict__ kbuf,
    const unsigned short* __restrict__ vbuf, const float* __restrict__ relb,
    unsigned short* __restrict__ ao) {
  __shared__ alignas(16) unsigned short qs[64 * 40];
  __shared__ alignas(16) unsigned short ks[64 * 40];
  __shared__ alignas(16) unsigned short vT[32 * 72];  // [d][tok]
  __shared__ alignas(16) unsigned short ps[64 * 72];  // softmax probs

  const int tid = threadIdx.x;
  const int lane = tid & 31;
  const int wv = tid >> 5;
  const int winId = blockIdx.x >> 3;
  const int head = blockIdx.x & 7;
  const int wloc = winId & 255;
  const size_t base = ((size_t)winId * 8 + head) * (64 * 32);

#if HAVE_TDM
  if (wv == 0) {
    tdm_load_tile_bf16(qbuf + base, lds_addr32(&qs[0]), 32, 64);
    tdm_load_tile_bf16(kbuf + base, lds_addr32(&ks[0]), 32, 64);
  }
#else
  for (int p = tid; p < 256; p += 128) {
    const int row = p >> 2;
    const int kq = (p & 3) * 8;
    *(u32x4*)&qs[row * 40 + kq] = *(const u32x4*)(qbuf + base + row * 32 + kq);
    *(u32x4*)&ks[row * 40 + kq] = *(const u32x4*)(kbuf + base + row * 32 + kq);
  }
#endif
  for (int p = tid; p < 2048; p += 128)
    vT[(p & 31) * 72 + (p >> 5)] = vbuf[base + p];
#if HAVE_TDM
  if (wv == 0) WAIT_TENSORCNT0();
#endif
  __syncthreads();

  const int m0 = wv * 16;
  const int hl = lane >> 4;
  const int nl = lane & 15;
  const bf16x16 aq = load_fragA(qs, m0, 0, 40, lane);
  floatx8 sc[4];
  #pragma unroll
  for (int ni = 0; ni < 4; ++ni) {
    const bf16x16 bk = load_fragB(ks, ni * 16, 0, 40, lane);
    floatx8 z = {};
    sc[ni] = wmma_bf16(aq, bk, z);
  }

  const int wh = wloc >> 4, ww = wloc & 15;
  #pragma unroll
  for (int ni = 0; ni < 4; ++ni) {
    #pragma unroll
    for (int r = 0; r < 8; ++r) {
      const int M = m0 + r + hl * 8;        // query token
      const int N = ni * 16 + nl;           // key token
      const int ia = M >> 3, ja = M & 7, ib = N >> 3, jb = N & 7;
      const int rpi = (ia - ib + 7) * 15 + (ja - jb + 7);
      float add = relb[rpi * 8 + head];
      const int ra = rgn((wh << 3) + ia) * 3 + rgn((ww << 3) + ja);
      const int rb = rgn((wh << 3) + ib) * 3 + rgn((ww << 3) + jb);
      if (ra != rb) add -= 100.0f;
      sc[ni][r] += add;
    }
  }

  #pragma unroll
  for (int r = 0; r < 8; ++r) {
    float mx = fmaxf(fmaxf(sc[0][r], sc[1][r]), fmaxf(sc[2][r], sc[3][r]));
    #pragma unroll
    for (int o = 8; o; o >>= 1) mx = fmaxf(mx, __shfl_xor(mx, o, 32));
    float sum = 0.0f;
    #pragma unroll
    for (int ni = 0; ni < 4; ++ni) {
      const float e = __expf(sc[ni][r] - mx);
      sc[ni][r] = e;
      sum += e;
    }
    #pragma unroll
    for (int o = 8; o; o >>= 1) sum += __shfl_xor(sum, o, 32);
    const float inv = 1.0f / sum;
    #pragma unroll
    for (int ni = 0; ni < 4; ++ni) sc[ni][r] *= inv;
  }

  #pragma unroll
  for (int ni = 0; ni < 4; ++ni)
    #pragma unroll
    for (int r = 0; r < 8; ++r)
      ps[(m0 + r + hl * 8) * 72 + ni * 16 + nl] = f2u16(sc[ni][r]);
  __syncthreads();

  floatx8 ov[2] = {};
  #pragma unroll
  for (int kk = 0; kk < 2; ++kk) {
    const bf16x16 ap = load_fragA(ps, m0, kk * 32, 72, lane);
    #pragma unroll
    for (int ni = 0; ni < 2; ++ni) {
      const bf16x16 bv = load_fragB(vT, ni * 16, kk * 32, 72, lane);
      ov[ni] = wmma_bf16(ap, bv, ov[ni]);
    }
  }
  #pragma unroll
  for (int ni = 0; ni < 2; ++ni)
    #pragma unroll
    for (int r = 0; r < 8; ++r) {
      const int M = m0 + r + hl * 8;
      ao[((size_t)winId * 64 + M) * 256 + head * 32 + ni * 16 + nl] = f2u16(ov[ni][r]);
    }
}

// ---------------------------------------------------------------------------
extern "C" void kernel_launch(void* const* d_in, const int* in_sizes, int n_in,
                              void* d_out, int out_size, void* d_ws, size_t ws_size,
                              hipStream_t stream) {
  (void)in_sizes; (void)n_in; (void)out_size; (void)ws_size;
  const float* x      = (const float*)d_in[0];
  const float* ln1_w  = (const float*)d_in[1];
  const float* ln1_b  = (const float*)d_in[2];
  const float* qkv_w  = (const float*)d_in[3];
  const float* qkv_b  = (const float*)d_in[4];
  const float* rel_b  = (const float*)d_in[5];
  const float* proj_w = (const float*)d_in[6];
  const float* proj_b = (const float*)d_in[7];
  const float* ln2_w  = (const float*)d_in[8];
  const float* ln2_b  = (const float*)d_in[9];
  const float* fc1_w  = (const float*)d_in[10];
  const float* fc1_b  = (const float*)d_in[11];
  const float* fc2_w  = (const float*)d_in[12];
  const float* fc2_b  = (const float*)d_in[13];

  char* ws = (char*)d_ws;
  size_t off = 0;
  auto take = [&](size_t bytes) {
    char* p = ws + off;
    off = (off + bytes + 255) & ~(size_t)255;
    return p;
  };

  unsigned short* wqkv  = (unsigned short*)take((size_t)768 * 256 * 2);
  unsigned short* wproj = (unsigned short*)take((size_t)256 * 256 * 2);
  unsigned short* wfc1  = (unsigned short*)take((size_t)1024 * 256 * 2);
  unsigned short* wfc2  = (unsigned short*)take((size_t)256 * 1024 * 2);
  unsigned short* win   = (unsigned short*)take((size_t)NTOKENS * 256 * 2);
  unsigned short* qb    = (unsigned short*)take((size_t)NTOKENS * 256 * 2 * 3);
  unsigned short* kb    = qb + (size_t)NTOKENS * 256;
  unsigned short* vb    = kb + (size_t)NTOKENS * 256;
  unsigned short* ao    = (unsigned short*)take((size_t)NTOKENS * 256 * 2);
  float*          tb    = (float*)take((size_t)NTOKENS * 256 * 4);
  unsigned short* h2    = (unsigned short*)take((size_t)NTOKENS * 1024 * 2);
  unsigned short* xn    = win;   // win is dead after the QKV GEMM: reuse for LN2 output

  cvt_bf16_kernel<<<(768 * 256 + 255) / 256, 256, 0, stream>>>(qkv_w, wqkv, 768 * 256);
  cvt_bf16_kernel<<<(256 * 256 + 255) / 256, 256, 0, stream>>>(proj_w, wproj, 256 * 256);
  cvt_bf16_kernel<<<(1024 * 256 + 255) / 256, 256, 0, stream>>>(fc1_w, wfc1, 1024 * 256);
  cvt_bf16_kernel<<<(256 * 1024 + 255) / 256, 256, 0, stream>>>(fc2_w, wfc2, 256 * 1024);

  ln1_window_kernel<<<NTOKENS / 8, 256, 0, stream>>>(x, ln1_w, ln1_b, win);

  gemm_wmma_kernel<0><<<dim3(6, 1024), 256, 0, stream>>>(win, wqkv, qkv_b,
                                                         nullptr, qb, nullptr);

  attn_kernel<<<NWINDOWS * 8, 128, 0, stream>>>(qb, kb, vb, rel_b, ao);

  gemm_wmma_kernel<1><<<dim3(2, 1024), 256, 0, stream>>>(ao, wproj, proj_b,
                                                         tb, nullptr, x);

  ln2_kernel<<<NTOKENS / 8, 256, 0, stream>>>(tb, ln2_w, ln2_b, xn);

  gemm_wmma_kernel<2><<<dim3(8, 1024), 256, 0, stream>>>(xn, wfc1, fc1_b,
                                                         nullptr, h2, nullptr);

  gemm_wmma_kernel<3><<<dim3(2, 1024), 256, 0, stream>>>(h2, wfc2, fc2_b,
                                                         (float*)d_out, nullptr, tb);
}
